// Qwen3Model_69071664054491
// MI455X (gfx1250) — compile-verified, matrix operand order checked by reading, not run on hardware
//
#include <hip/hip_runtime.h>

// ---------------- model dims ----------------
constexpr int S_ = 2048, D_ = 2048, H_ = 16, KV_ = 8, HD_ = 128, INTER_ = 6144, L_ = 2;
constexpr float EPS_ = 1e-6f;
constexpr float QSCALE_ = 0.08838834764831845f; // 1/sqrt(128)

// ---------------- vector types ----------------
typedef __attribute__((ext_vector_type(16))) __bf16 v16bf;
typedef __attribute__((ext_vector_type(8)))  __bf16 v8bf;
typedef __attribute__((ext_vector_type(8)))  float  v8f;
typedef __attribute__((ext_vector_type(4)))  float  v4f;

__device__ __forceinline__ __bf16 f2b(float f) { return static_cast<__bf16>(f); }

__device__ __forceinline__ v16bf cat8(v8bf lo, v8bf hi) {
  return __builtin_shufflevector(lo, hi, 0,1,2,3,4,5,6,7,8,9,10,11,12,13,14,15);
}

__device__ __forceinline__ v8bf cvt8(v4f a, v4f b) {
  v8f t = __builtin_shufflevector(a, b, 0,1,2,3,4,5,6,7);
  return __builtin_convertvector(t, v8bf); // hardware f32->bf16 packed converts
}

__device__ __forceinline__ v8f wmma_bf16(v16bf a, v16bf b, v8f c) {
  return __builtin_amdgcn_wmma_f32_16x16x32_bf16(false, a, false, b, (short)0, c, false, false);
}

// wave-local LDS sync (same wave produced + consumes; LDS ops are in-order per wave)
__device__ __forceinline__ void lds_wave_sync() {
  asm volatile("s_wait_dscnt 0" ::: "memory");
}

// ---------------- fp32 -> bf16 weight pre-conversion ----------------
__global__ __launch_bounds__(256) void w2bf_kernel(const float* __restrict__ W,
                                                   __bf16* __restrict__ O, size_t n8) {
  size_t i = (size_t)blockIdx.x * 256 + threadIdx.x;
  if (i < n8) {
    const v4f* p = (const v4f*)(W + i * 8);
    *(v8bf*)(O + i * 8) = cvt8(p[0], p[1]);
  }
}

// ---------------- embedding gather ----------------
__global__ __launch_bounds__(256) void embed_kernel(const int* __restrict__ ids,
                                                    const float* __restrict__ emb,
                                                    float* __restrict__ h) {
  int s = blockIdx.y;
  int d = blockIdx.x * 256 + threadIdx.x;
  h[(size_t)s * D_ + d] = emb[(size_t)ids[s] * D_ + d];
}

// ---------------- RMSNorm over D, fp32 -> bf16 ----------------
__global__ __launch_bounds__(256) void rmsnorm_kernel(const float* __restrict__ X,
                                                      const float* __restrict__ scale,
                                                      __bf16* __restrict__ Y) {
  __shared__ float red[256];
  int s = blockIdx.x, tid = threadIdx.x;
  const float* xp = X + (size_t)s * D_;
  float ss = 0.f;
  for (int i = tid; i < D_; i += 256) { float v = xp[i]; ss += v * v; }
  red[tid] = ss; __syncthreads();
  for (int o = 128; o > 0; o >>= 1) {
    if (tid < o) red[tid] += red[tid + o];
    __syncthreads();
  }
  float r = rsqrtf(red[0] * (1.0f / D_) + EPS_);
  for (int i = tid; i < D_; i += 256) Y[(size_t)s * D_ + i] = f2b(xp[i] * r * scale[i]);
}

// ---------------- generic NT GEMM: C[M,N] = A[M,K](bf16) * W[N,K]^T ----------------
// MODE 0: C = acc (f32)    MODE 1: C = Res + acc (f32, fused residual)
// WBF  0: W is f32 (convert in-register)   WBF 1: W is bf16 (pre-converted)
template <int MODE, int WBF>
__global__ __launch_bounds__(256) void gemm_nt(const __bf16* __restrict__ A,
                                               const void* __restrict__ Wv,
                                               float* __restrict__ C,
                                               const float* __restrict__ Res,
                                               int M, int N, int K) {
  __shared__ __bf16 As[2][128 * 40]; // 32 bf16/row + 8 pad -> conflict-free frag reads
  const int tid = threadIdx.x;
  const int wave = tid >> 5, lane = tid & 31;
  const int wm = (wave >> 2) * 64;   // 0 or 64
  const int wn = (wave & 3) * 32;    // 0..96
  const int bm = blockIdx.y * 128;
  const int bn = blockIdx.x * 128;
  const int lrow = lane & 15;
  const int khalf = (lane >> 4) * 8; // 0 or 8
  const int rowhi = (lane >> 4) * 8;
  const int arow = tid >> 1, aseg = (tid & 1) * 16;

  v8f acc[4][2];
#pragma unroll
  for (int mi = 0; mi < 4; ++mi)
#pragma unroll
    for (int ni = 0; ni < 2; ++ni)
#pragma unroll
      for (int i = 0; i < 8; ++i) acc[mi][ni][i] = 0.f;

  // prologue: stage tile 0
  {
    const __bf16* src = A + (size_t)(bm + arow) * K + aseg;
    v8bf x0 = *(const v8bf*)src;
    v8bf x1 = *(const v8bf*)(src + 8);
    __bf16* dst = &As[0][arow * 40 + aseg];
    *(v8bf*)dst = x0; *(v8bf*)(dst + 8) = x1;
  }
  __syncthreads();

  const int nk = K >> 5;
  for (int kt = 0; kt < nk; ++kt) {
    const int k0 = kt * 32;
    const int cur = kt & 1;
    const bool more = (kt + 1) < nk;

    // issue next A tile's global loads early
    v8bf nx0, nx1;
    if (more) {
      const __bf16* src = A + (size_t)(bm + arow) * K + (k0 + 32) + aseg;
      nx0 = *(const v8bf*)src;
      nx1 = *(const v8bf*)(src + 8);
    }

    // B fragments (global)
    v16bf bfrag[2];
#pragma unroll
    for (int ni = 0; ni < 2; ++ni) {
      int n = bn + wn + ni * 16 + lrow;
      if (WBF) {
        const __bf16* bp = (const __bf16*)Wv + (size_t)n * K + k0 + khalf;
        bfrag[ni] = cat8(*(const v8bf*)bp, *(const v8bf*)(bp + 16));
      } else {
        const float* bp = (const float*)Wv + (size_t)n * K + k0 + khalf;
        v4f b0 = *(const v4f*)bp;        v4f b1 = *(const v4f*)(bp + 4);
        v4f b2 = *(const v4f*)(bp + 16); v4f b3 = *(const v4f*)(bp + 20);
        bfrag[ni] = cat8(cvt8(b0, b1), cvt8(b2, b3));
      }
    }
#pragma unroll
    for (int mi = 0; mi < 4; ++mi) {
      const __bf16* ap = &As[cur][(wm + mi * 16 + lrow) * 40 + khalf];
      v16bf af = cat8(*(const v8bf*)ap, *(const v8bf*)(ap + 16));
      acc[mi][0] = wmma_bf16(af, bfrag[0], acc[mi][0]);
      acc[mi][1] = wmma_bf16(af, bfrag[1], acc[mi][1]);
    }

    if (more) {
      __bf16* dst = &As[cur ^ 1][arow * 40 + aseg];
      *(v8bf*)dst = nx0; *(v8bf*)(dst + 8) = nx1;
    }
    __syncthreads();
  }

#pragma unroll
  for (int mi = 0; mi < 4; ++mi)
#pragma unroll
    for (int ni = 0; ni < 2; ++ni) {
      int n = bn + wn + ni * 16 + lrow;
#pragma unroll
      for (int i = 0; i < 8; ++i) {
        int m = bm + wm + mi * 16 + rowhi + i;
        size_t idx = (size_t)m * N + n;
        float v = acc[mi][ni][i];
        if (MODE == 1) C[idx] = Res[idx] + v; else C[idx] = v;
      }
    }
}

// ---------------- per-head RMSNorm + RoPE, [S][NH][HD] f32 -> [NH][S][HD] bf16 ----------------
__global__ __launch_bounds__(256) void qknorm_rope_kernel(const float* __restrict__ X,
                                                          const float* __restrict__ nscale,
                                                          const float* __restrict__ cosT,
                                                          const float* __restrict__ sinT,
                                                          __bf16* __restrict__ Y,
                                                          int NH, float outScale) {
  int wid = (blockIdx.x * blockDim.x + threadIdx.x) >> 5; // one wave per (s, head) row
  int lane = threadIdx.x & 31;
  int s = wid / NH, hh = wid % NH;
  const float* xp = X + (size_t)wid * HD_;
  float v0 = xp[lane], v1 = xp[lane + 32], v2 = xp[lane + 64], v3 = xp[lane + 96];
  float ss = v0 * v0 + v1 * v1 + v2 * v2 + v3 * v3;
#pragma unroll
  for (int o = 16; o > 0; o >>= 1) ss += __shfl_xor(ss, o, 32);
  float r = rsqrtf(ss * (1.0f / HD_) + EPS_);

  const float* cs = cosT + (size_t)s * HD_;
  const float* sn = sinT + (size_t)s * HD_;
  __bf16* yp = Y + ((size_t)hh * S_ + s) * HD_;
  float xs[4]  = {v0, v1, v2, v3};
  float prt[4] = {v2, v3, v0, v1};           // partner element (i ^ 64)
  int   idxs[4] = {lane, lane + 32, lane + 64, lane + 96};
#pragma unroll
  for (int e = 0; e < 4; ++e) {
    int i = idxs[e];
    float xn  = xs[e]  * r * nscale[i];
    float xpn = prt[e] * r * nscale[i ^ 64];
    float sign = (i < 64) ? -1.f : 1.f;      // rotated = concat(-x2, x1)
    float o = xn * cs[i] + sign * xpn * sn[i];
    yp[i] = f2b(o * outScale);
  }
}

// ---------------- V transpose: [S][KV*HD] f32 -> [KV*HD][S] bf16 (LDS tiled) ----------------
__global__ __launch_bounds__(256) void vtrans_kernel(const float* __restrict__ V,
                                                     __bf16* __restrict__ Vt) {
  __shared__ float t[32][33];
  int s0 = blockIdx.x * 32, c0 = blockIdx.y * 32;
  int tid = threadIdx.x;
  int cc = tid & 31, rr = tid >> 5; // 8 rows per pass
#pragma unroll
  for (int i = 0; i < 4; ++i) {
    int row = rr + i * 8;
    t[row][cc] = V[(size_t)(s0 + row) * (KV_ * HD_) + c0 + cc];
  }
  __syncthreads();
#pragma unroll
  for (int i = 0; i < 4; ++i) {
    int crow = rr + i * 8;
    Vt[(size_t)(c0 + crow) * S_ + s0 + cc] = f2b(t[cc][crow]);
  }
}

// ---------------- flash attention: one wave per 16-query tile ----------------
// Q: [H][S][HD] bf16 (pre-scaled by 1/sqrt(HD)), K: [KV][S][HD] bf16, Vt: [KV][HD][S] bf16
// Out: [S][H*HD] bf16
__global__ __launch_bounds__(256) void attn_kernel(const __bf16* __restrict__ Q,
                                                   const __bf16* __restrict__ Kc,
                                                   const __bf16* __restrict__ Vt,
                                                   __bf16* __restrict__ Out) {
  __shared__ float  sS[8][16][33];
  __shared__ __bf16 sP[8][16][32];
  __shared__ float  sAlpha[8][16];
  __shared__ float  sInvL[8][16];

  const int wave = threadIdx.x >> 5, lane = threadIdx.x & 31;
  const int h = blockIdx.y, kv = h >> 1; // GROUP = H/KV = 2
  const int q0 = (blockIdx.x * 8 + wave) * 16;
  const int lrow = lane & 15;
  const int khalf = (lane >> 4) * 8;
  const int rowhi = (lane >> 4) * 8;

  // preload the 16x128 Q tile as 4 A-fragments
  const __bf16* qbase = Q + ((size_t)h * S_ + q0) * HD_;
  v16bf qf[4];
#pragma unroll
  for (int kk = 0; kk < 4; ++kk) {
    const __bf16* p = qbase + (size_t)lrow * HD_ + kk * 32 + khalf;
    qf[kk] = cat8(*(const v8bf*)p, *(const v8bf*)(p + 16));
  }

  v8f acc[8];
#pragma unroll
  for (int t = 0; t < 8; ++t)
#pragma unroll
    for (int i = 0; i < 8; ++i) acc[t][i] = 0.f;

  float m_r = -INFINITY, l_r = 0.f; // valid in lanes 0..15 (one row each)
  const int nkt = (q0 + 47) / 32;   // key tiles of 32 covering keys 0..q0+15

  for (int kt = 0; kt < nkt; ++kt) {
    const int kb0 = kt * 32;

    // ---- preload all 8 K fragments, then chain 8 WMMAs ----
    v16bf kf[8];
#pragma unroll
    for (int ns = 0; ns < 2; ++ns) {
      int key = kb0 + ns * 16 + lrow;
      const __bf16* kp = Kc + ((size_t)kv * S_ + key) * HD_ + khalf;
#pragma unroll
      for (int kk = 0; kk < 4; ++kk)
        kf[ns * 4 + kk] = cat8(*(const v8bf*)(kp + kk * 32), *(const v8bf*)(kp + kk * 32 + 16));
    }
#pragma unroll
    for (int ns = 0; ns < 2; ++ns) {
      v8f sc;
#pragma unroll
      for (int i = 0; i < 8; ++i) sc[i] = 0.f;
#pragma unroll
      for (int kk = 0; kk < 4; ++kk) sc = wmma_bf16(qf[kk], kf[ns * 4 + kk], sc);
#pragma unroll
      for (int i = 0; i < 8; ++i) sS[wave][rowhi + i][ns * 16 + lrow] = sc[i];
    }
    lds_wave_sync();

    // ---- online softmax, row r owned by lane r ----
    if (lane < 16) {
      int r = lane, qidx = q0 + r;
      float sv[32];
      float mx = -INFINITY;
#pragma unroll
      for (int c = 0; c < 32; ++c) {
        float x = sS[wave][r][c];
        if (kb0 + c > qidx) x = -INFINITY; // causal mask
        sv[c] = x;
        mx = fmaxf(mx, x);
      }
      float mnew = fmaxf(m_r, mx);
      float al, sum = 0.f;
      if (mnew == -INFINITY) {
        al = 1.f;
#pragma unroll
        for (int c = 0; c < 32; ++c) sP[wave][r][c] = f2b(0.f);
      } else {
        al = __expf(m_r - mnew); // m_r==-inf -> 0
#pragma unroll
        for (int c = 0; c < 32; ++c) {
          float p = (sv[c] == -INFINITY) ? 0.f : __expf(sv[c] - mnew);
          sum += p;
          sP[wave][r][c] = f2b(p);
        }
      }
      l_r = al * l_r + sum;
      m_r = mnew;
      sAlpha[wave][r] = al;
    }
    lds_wave_sync();

    // ---- preload 8 V fragments + P fragment, rescale, chain 8 WMMAs ----
    float al[8];
#pragma unroll
    for (int i = 0; i < 8; ++i) al[i] = sAlpha[wave][rowhi + i];
    v16bf pf;
    {
      const __bf16* pp = &sP[wave][lrow][khalf];
      pf = cat8(*(const v8bf*)pp, *(const v8bf*)(pp + 16));
    }
    v16bf vf[8];
#pragma unroll
    for (int t = 0; t < 8; ++t) {
      int hd = t * 16 + lrow;
      const __bf16* vp = Vt + ((size_t)kv * HD_ + hd) * S_ + kb0 + khalf;
      vf[t] = cat8(*(const v8bf*)vp, *(const v8bf*)(vp + 16));
    }
#pragma unroll
    for (int t = 0; t < 8; ++t) {
#pragma unroll
      for (int i = 0; i < 8; ++i) acc[t][i] *= al[i];
      acc[t] = wmma_bf16(pf, vf[t], acc[t]);
    }
  }

  if (lane < 16) sInvL[wave][lane] = 1.f / l_r; // causal => l_r > 0
  lds_wave_sync();
  float il[8];
#pragma unroll
  for (int i = 0; i < 8; ++i) il[i] = sInvL[wave][rowhi + i];
#pragma unroll
  for (int t = 0; t < 8; ++t) {
    int hd = t * 16 + lrow;
#pragma unroll
    for (int i = 0; i < 8; ++i) {
      int q = q0 + rowhi + i;
      Out[(size_t)q * (H_ * HD_) + h * HD_ + hd] = f2b(acc[t][i] * il[i]);
    }
  }
}

// ---------------- SiLU(gate) * up -> bf16 ----------------
__global__ __launch_bounds__(256) void silu_mul_kernel(const float* __restrict__ G,
                                                       const float* __restrict__ U,
                                                       __bf16* __restrict__ T) {
  size_t idx = (size_t)blockIdx.x * 256 + threadIdx.x;
  float g = G[idx];
  float t = g / (1.f + __expf(-g)) * U[idx];
  T[idx] = f2b(t);
}

// ---------------- host orchestration ----------------
extern "C" void kernel_launch(void* const* d_in, const int* in_sizes, int n_in,
                              void* d_out, int out_size, void* d_ws, size_t ws_size,
                              hipStream_t stream) {
  (void)in_sizes; (void)n_in; (void)out_size;
  const int*   ids   = (const int*)d_in[0];
  // d_in[1] = mask (causal; computed analytically in-kernel)
  const float* cosT  = (const float*)d_in[2];
  const float* sinT  = (const float*)d_in[3];
  const float* emb   = (const float*)d_in[4];
  const float* q_w   = (const float*)d_in[5];
  const float* k_w   = (const float*)d_in[6];
  const float* v_w   = (const float*)d_in[7];
  const float* o_w   = (const float*)d_in[8];
  const float* qn    = (const float*)d_in[9];
  const float* kn    = (const float*)d_in[10];
  const float* ln1   = (const float*)d_in[11];
  const float* ln2   = (const float*)d_in[12];
  const float* g_w   = (const float*)d_in[13];
  const float* u_w   = (const float*)d_in[14];
  const float* dn_w  = (const float*)d_in[15];

  float* h = (float*)d_out; // residual stream lives in the output buffer

  char* ws = (char*)d_ws;
  size_t off = 0;
  auto alloc = [&](size_t bytes) -> void* {
    void* p = ws + off;
    off += (bytes + 255) & ~(size_t)255;
    return p;
  };
  __bf16* xb    = (__bf16*)alloc((size_t)S_ * D_ * 2);
  float*  q32   = (float*) alloc((size_t)S_ * H_ * HD_ * 4);
  float*  k32   = (float*) alloc((size_t)S_ * KV_ * HD_ * 4);
  float*  v32   = (float*) alloc((size_t)S_ * KV_ * HD_ * 4);
  __bf16* qb    = (__bf16*)alloc((size_t)H_ * S_ * HD_ * 2);
  __bf16* kb    = (__bf16*)alloc((size_t)KV_ * S_ * HD_ * 2);
  __bf16* vtb   = (__bf16*)alloc((size_t)KV_ * HD_ * S_ * 2);
  __bf16* attnb = (__bf16*)alloc((size_t)S_ * H_ * HD_ * 2);
  float*  g32   = (float*) alloc((size_t)S_ * INTER_ * 4);
  float*  u32   = (float*) alloc((size_t)S_ * INTER_ * 4);
  __bf16* tb    = (__bf16*)alloc((size_t)S_ * INTER_ * 2);

  // pre-converted bf16 weights (full L-stacked arrays), if workspace allows
  const size_t nq = (size_t)L_ * D_ * D_;          // q_w / o_w element counts
  const size_t nk = (size_t)L_ * KV_ * HD_ * D_;   // k_w / v_w
  const size_t ng = (size_t)L_ * INTER_ * D_;      // gate / up / down
  const size_t wbytes = (nq * 2 + nk * 2 + ng * 3) * 2 + 7 * 256;
  const bool convw = (ws_size >= off + wbytes);
  __bf16 *qwb = nullptr, *kwb = nullptr, *vwb = nullptr, *owb = nullptr,
         *gwb = nullptr, *uwb = nullptr, *dwb = nullptr;
  if (convw) {
    qwb = (__bf16*)alloc(nq * 2); kwb = (__bf16*)alloc(nk * 2);
    vwb = (__bf16*)alloc(nk * 2); owb = (__bf16*)alloc(nq * 2);
    gwb = (__bf16*)alloc(ng * 2); uwb = (__bf16*)alloc(ng * 2);
    dwb = (__bf16*)alloc(ng * 2);
    auto conv = [&](const float* src, __bf16* dst, size_t n) {
      size_t n8 = n / 8;
      w2bf_kernel<<<(unsigned)((n8 + 255) / 256), 256, 0, stream>>>(src, dst, n8);
    };
    conv(q_w, qwb, nq); conv(k_w, kwb, nk); conv(v_w, vwb, nk); conv(o_w, owb, nq);
    conv(g_w, gwb, ng); conv(u_w, uwb, ng); conv(dn_w, dwb, ng);
  }

  embed_kernel<<<dim3(D_ / 256, S_), 256, 0, stream>>>(ids, emb, h);

  auto gemm = [&](int mode, const __bf16* A, const float* Wf, const __bf16* Wb,
                  float* C, const float* Res, int M, int N, int K) {
    dim3 grid(N / 128, M / 128);
    if (convw) {
      if (mode == 0) gemm_nt<0, 1><<<grid, 256, 0, stream>>>(A, Wb, C, Res, M, N, K);
      else           gemm_nt<1, 1><<<grid, 256, 0, stream>>>(A, Wb, C, Res, M, N, K);
    } else {
      if (mode == 0) gemm_nt<0, 0><<<grid, 256, 0, stream>>>(A, Wf, C, Res, M, N, K);
      else           gemm_nt<1, 0><<<grid, 256, 0, stream>>>(A, Wf, C, Res, M, N, K);
    }
  };

  for (int l = 0; l < L_; ++l) {
    const size_t oq = (size_t)l * D_ * D_;
    const size_t ok = (size_t)l * KV_ * HD_ * D_;
    const size_t og = (size_t)l * INTER_ * D_;

    // attention block
    rmsnorm_kernel<<<S_, 256, 0, stream>>>(h, ln1 + (size_t)l * D_, xb);
    gemm(0, xb, q_w + oq, qwb ? qwb + oq : nullptr, q32, nullptr, S_, D_, D_);
    gemm(0, xb, k_w + ok, kwb ? kwb + ok : nullptr, k32, nullptr, S_, KV_ * HD_, D_);
    gemm(0, xb, v_w + ok, vwb ? vwb + ok : nullptr, v32, nullptr, S_, KV_ * HD_, D_);
    qknorm_rope_kernel<<<(S_ * H_) / 8, 256, 0, stream>>>(q32, qn + (size_t)l * HD_, cosT, sinT, qb, H_, QSCALE_);
    qknorm_rope_kernel<<<(S_ * KV_) / 8, 256, 0, stream>>>(k32, kn + (size_t)l * HD_, cosT, sinT, kb, KV_, 1.0f);
    vtrans_kernel<<<dim3(S_ / 32, (KV_ * HD_) / 32), 256, 0, stream>>>(v32, vtb);
    attn_kernel<<<dim3(S_ / 128, H_), 256, 0, stream>>>(qb, kb, vtb, attnb);
    gemm(1, attnb, o_w + oq, owb ? owb + oq : nullptr, h, h, S_, D_, H_ * HD_);

    // MLP block
    rmsnorm_kernel<<<S_, 256, 0, stream>>>(h, ln2 + (size_t)l * D_, xb);
    gemm(0, xb, g_w + og, gwb ? gwb + og : nullptr, g32, nullptr, S_, INTER_, D_);
    gemm(0, xb, u_w + og, uwb ? uwb + og : nullptr, u32, nullptr, S_, INTER_, D_);
    silu_mul_kernel<<<(unsigned)((size_t)S_ * INTER_ / 256), 256, 0, stream>>>(g32, u32, tb);
    gemm(1, tb, dn_w + og, dwb ? dwb + og : nullptr, h, h, S_, D_, INTER_);
  }
}
